// OrthoLinear_40381282517453
// MI455X (gfx1250) — matrix operand order repaired and checked by reading, not compiled, not run on hardware
//
#include <hip/hip_runtime.h>

// ---- vector types ----
typedef __attribute__((ext_vector_type(16))) _Float16 v16h;
typedef __attribute__((ext_vector_type(8)))  float    v8f;
// may_alias variants for vector memory ops that overlap scalar fp16 LDS RMWs.
typedef _Float16 v8h_a  __attribute__((ext_vector_type(8), may_alias));
typedef int      int4_a __attribute__((ext_vector_type(4), may_alias));
typedef float    f4_a   __attribute__((ext_vector_type(4), may_alias));

#define TOKENS 256
#define IN_F   8192
#define OUT_F  8192

#define BM 128            // token tile
#define BN 64             // out-feature tile
#define BK 32             // K step (WMMA f16 K)
#define LDS_STRIDE 40     // 32 halves + 8 pad -> 80B rows (16B aligned, conflict-free)

__global__ __launch_bounds__(256, 1)
void ortho_linear_wmma_kernel(const float* __restrict__ x,
                              const int* __restrict__ base_packed,
                              const float* __restrict__ scales,
                              const _Float16* __restrict__ ovals,
                              const int* __restrict__ oidx,
                              const int* __restrict__ optr,
                              float* __restrict__ out)
{
    // double-buffered tiles
    __shared__ __align__(16) _Float16 lds_x[2 * BM * LDS_STRIDE];   // 20 KB
    __shared__ __align__(16) _Float16 lds_w[2 * BN * LDS_STRIDE];   // 10 KB

    const int t    = threadIdx.x;
    const int lane = t & 31;
    const int wv   = t >> 5;      // wave 0..7
    const int wm   = wv >> 1;     // 0..3 (M direction, 32 rows each)
    const int wn   = wv & 1;      // 0..1 (N direction, 32 cols each)

    const int m0 = blockIdx.y * BM;
    const int n0 = blockIdx.x * BN;

    // --- staging roles ---
    // x: thread owns (row rx, 16-wide half hx): 128 rows x 2 halves = 256 threads
    const int rx = t >> 1;
    const int hx = t & 1;
    // w: thread owns (row rw, 8-wide quarter q): 64 rows x 4 quarters = 256 threads
    const int rw = t >> 2;
    const int q  = t & 3;

    const float* xrow = x + (size_t)(m0 + rx) * IN_F + hx * 16;
    const int*   wrow = base_packed + (size_t)(n0 + rw) * (IN_F / 2) + q * 4;
    const float  scale = scales[n0 + rw];

    // CSR cursor for residual of out-row (n0 + rw); indices sorted ascending.
    int       cur   = optr[n0 + rw];
    const int end   = optr[n0 + rw + 1];
    int       nextc = (cur < end) ? oidx[cur] : 0x7fffffff;

    v8f zero = {};
    v8f acc[2][2];
#pragma unroll
    for (int mi = 0; mi < 2; ++mi)
#pragma unroll
        for (int ni = 0; ni < 2; ++ni)
            acc[mi][ni] = zero;

    // fragment addressing (CDNA5 16-bit WMMA layouts, wave32)
    const int fr = lane & 15;             // M (A) or N (B) index
    const int kh = (lane >> 4) * 8;       // A: K groups {0..7,16..23} vs {8..15,24..31}
    const int kb = (lane >> 4) * 16;      // unused; B reads also use kh (same K groups as A)

    // ---- stage one K-tile into LDS buffer `buf` ----
    auto stage = [&](int k0, int buf) {
        _Float16* bx = lds_x + buf * (BM * LDS_STRIDE);
        _Float16* bw = lds_w + buf * (BN * LDS_STRIDE);

        // x tile: fp32 -> fp16 (16 floats per thread)
        {
            const f4_a* p = (const f4_a*)(xrow + k0);
            f4_a q0 = p[0], q1 = p[1], q2 = p[2], q3 = p[3];
            float qs[16];
#pragma unroll
            for (int i = 0; i < 4; ++i) {
                qs[i] = q0[i]; qs[i + 4] = q1[i]; qs[i + 8] = q2[i]; qs[i + 12] = q3[i];
            }
            v8h_a lo, hi;
#pragma unroll
            for (int i = 0; i < 8; ++i) {
                lo[i] = (_Float16)qs[i];
                hi[i] = (_Float16)qs[i + 8];
            }
            *(v8h_a*)&bx[rx * LDS_STRIDE + hx * 16]     = lo;
            *(v8h_a*)&bx[rx * LDS_STRIDE + hx * 16 + 8] = hi;
        }

        // W tile: int4 nibble dequant -> fp16 (4 packed int32 -> 8 weights)
        {
            const int4_a* p = (const int4_a*)(wrow + (k0 >> 1));
            int4_a b4 = p[0];
            v8h_a wv8;
#pragma unroll
            for (int i = 0; i < 4; ++i) {
                int b = b4[i];
                wv8[2 * i]     = (_Float16)((float)((b & 0xF) - 8) * scale);
                wv8[2 * i + 1] = (_Float16)((float)(((b >> 4) & 0xF) - 8) * scale);
            }
            *(v8h_a*)&bw[rw * LDS_STRIDE + q * 8] = wv8;
        }

        // CSR residual: monotone cursor, same-thread in-order LDS RMW.
        // Thread applies entries inside its 8-wide quarter [k0+q*8, k0+q*8+8);
        // entries below that were applied by sibling threads in earlier steps.
        {
            const int lim = k0 + q * 8 + 8;
            while (nextc < lim) {
                if (nextc >= lim - 8) {
                    const int off = rw * LDS_STRIDE + (nextc - k0);
                    bw[off] = (_Float16)((float)bw[off] + (float)ovals[cur]);
                }
                ++cur;
                nextc = (cur < end) ? oidx[cur] : 0x7fffffff;
            }
        }

        // prefetch the K-tile we will stage next iteration
        if (k0 + BK < IN_F) {
            __builtin_prefetch((const void*)(wrow + ((k0 + BK) >> 1)), 0, 1);
            __builtin_prefetch((const void*)(xrow + (k0 + BK)), 0, 1);
        }
    };

    // ---- fragments + 4 WMMAs on LDS buffer `buf` ----
    auto compute = [&](int buf) {
        const _Float16* bx = lds_x + buf * (BM * LDS_STRIDE);
        const _Float16* bw = lds_w + buf * (BN * LDS_STRIDE);

        v16h afr[2];
#pragma unroll
        for (int mi = 0; mi < 2; ++mi) {
            const int mrow = wm * 32 + mi * 16 + fr;
            v8h_a alo = *(const v8h_a*)&bx[mrow * LDS_STRIDE + kh];
            v8h_a ahi = *(const v8h_a*)&bx[mrow * LDS_STRIDE + kh + 16];
#pragma unroll
            for (int i = 0; i < 8; ++i) { afr[mi][i] = alo[i]; afr[mi][i + 8] = ahi[i]; }
        }
        v16h bfr[2];
#pragma unroll
        for (int ni = 0; ni < 2; ++ni) {
            const int nrow = wn * 32 + ni * 16 + fr;
            v8h_a blo = *(const v8h_a*)&bw[nrow * LDS_STRIDE + kh];
            v8h_a bhi = *(const v8h_a*)&bw[nrow * LDS_STRIDE + kh + 16];
#pragma unroll
            for (int i = 0; i < 8; ++i) { bfr[ni][i] = blo[i]; bfr[ni][i + 8] = bhi[i]; }
        }
#pragma unroll
        for (int mi = 0; mi < 2; ++mi)
#pragma unroll
            for (int ni = 0; ni < 2; ++ni)
                acc[mi][ni] = __builtin_amdgcn_wmma_f32_16x16x32_f16(
                    false, afr[mi], false, bfr[ni],
                    (short)0, acc[mi][ni], false, false);
    };

    // ---- software-pipelined main loop: stage(k+1) overlaps compute(k) ----
    stage(0, 0);
    __syncthreads();
    int sel = 0;
    for (int k0 = BK; k0 < IN_F; k0 += BK) {
        stage(k0, sel ^ 1);
        compute(sel);
        __syncthreads();
        sel ^= 1;
    }
    compute(sel);

    // ---- epilogue: C/D layout -> global ----
    // acc element v: M = v (lanes 0-15) / v+8 (lanes 16-31), N = lane&15
    const int dm    = (lane >> 4) * 8;
    const int mbase = m0 + wm * 32;
    const int nbase = n0 + wn * 32;
#pragma unroll
    for (int mi = 0; mi < 2; ++mi)
#pragma unroll
        for (int ni = 0; ni < 2; ++ni) {
            const int o = nbase + ni * 16 + fr;
#pragma unroll
            for (int v = 0; v < 8; ++v) {
                const int tok = mbase + mi * 16 + dm + v;
                out[(size_t)tok * OUT_F + o] = acc[mi][ni][v];
            }
        }
}

extern "C" void kernel_launch(void* const* d_in, const int* in_sizes, int n_in,
                              void* d_out, int out_size, void* d_ws, size_t ws_size,
                              hipStream_t stream) {
    (void)in_sizes; (void)n_in; (void)out_size; (void)d_ws; (void)ws_size;
    const float*    x  = (const float*)d_in[0];
    const int*      bp = (const int*)d_in[1];
    const float*    sc = (const float*)d_in[2];
    const _Float16* ov = (const _Float16*)d_in[3];
    const int*      oi = (const int*)d_in[4];
    const int*      op = (const int*)d_in[5];
    float*          out = (float*)d_out;

    dim3 grid(OUT_F / BN, TOKENS / BM);   // 128 x 2 = 256 blocks
    ortho_linear_wmma_kernel<<<grid, 256, 0, stream>>>(x, bp, sc, ov, oi, op, out);
}
